// Block_523986010339
// MI455X (gfx1250) — matrix operand order repaired and checked by reading, not compiled, not run on hardware
//
#include <hip/hip_runtime.h>
#include <hip/hip_bf16.h>

// ---------------------------------------------------------------------------
// SegFormer block on gfx1250: all GEMMs via v_wmma_f32_16x16x32_f16 (wave32).
// B=8, N=16384 (=128x128), C=64, SR=8 -> Nr=256, HID=256, heads=1 (dh=64).
// Round 3: explicit double-buffered software pipelining in the K loops so
// global loads overlap the WMMA pipe (kills the s_wait_loadcnt 0x0 pattern).
// ---------------------------------------------------------------------------

typedef _Float16 h8   __attribute__((ext_vector_type(8)));
typedef _Float16 v16h __attribute__((ext_vector_type(16)));
typedef float    v8f  __attribute__((ext_vector_type(8)));

#define DEVI __device__ __forceinline__

DEVI int lane_id() { return (int)(threadIdx.x & 31); }

DEVI v8f zero8() {
  v8f z;
#pragma unroll
  for (int i = 0; i < 8; ++i) z[i] = 0.f;
  return z;
}

DEVI v16h frag_combine(h8 lo, h8 hi) {
  v16h r;
#pragma unroll
  for (int i = 0; i < 8; ++i) { r[i] = lo[i]; r[i + 8] = hi[i]; }
  return r;
}

// A fragment (16x32 f16, row-major). rowbase = &A[m = lane&15][k0].
// lanes 0-15 kb=0, lanes 16-31 kb=8; elems 0..7 -> k=kb..kb+7,
// elems 8..15 -> k=16+kb..16+kb+7  => two contiguous 16B loads.
DEVI v16h load_a_frag(const _Float16* rowbase) {
  const int kb = (lane_id() & 16) ? 8 : 0;
  h8 lo = *(const h8*)(rowbase + kb);
  h8 hi = *(const h8*)(rowbase + kb + 16);
  return frag_combine(lo, hi);
}

// B fragment (32x16 f16, column-major == weight W[N,K] row-major).
// colbase = &W[n = n0 + (lane&15)][k0]. kb = 0/8; elems 0..7 -> k = kb..kb+7, 8..15 -> k = 16+kb..16+kb+7.
template <typename P>
DEVI v16h load_b_frag(const P* colbase) {
  const int kb = (lane_id() & 16) ? 8 : 0;
  h8 lo = *(const h8*)(colbase + kb);
  h8 hi = *(const h8*)(colbase + kb + 16);
  return frag_combine(lo, hi);
}

DEVI v8f wmma16(v16h a, v16h b, v8f c) {
  // D = A(16x32) * B(32x16) + C, f32 accumulate
  return __builtin_amdgcn_wmma_f32_16x16x32_f16(false, a, false, b,
                                                (short)0, c, false, false);
}

// ---------------------------------------------------------------------------
// Weight conversion helpers (one-time, tiny)
// ---------------------------------------------------------------------------
__global__ void cvt_f16(const float* __restrict__ s, _Float16* __restrict__ d, int n) {
  int i = blockIdx.x * blockDim.x + threadIdx.x;
  if (i < n) d[i] = (_Float16)s[i];
}

// sr_w [64][64][8][8] -> srwT [n][k'] with k' = (i*8+j)*64 + c  (K=4096)
__global__ void srw_reorder(const float* __restrict__ s, _Float16* __restrict__ d) {
  int i = blockIdx.x * blockDim.x + threadIdx.x;
  if (i < 64 * 4096) {
    int n = i >> 12, k = i & 4095, sp = k >> 6, c = k & 63;
    d[i] = (_Float16)s[((size_t)(n * 64 + c)) * 64 + sp];
  }
}

// ---------------------------------------------------------------------------
// LayerNorm over C=64, one wave per token (lane handles 2 channels)
// ---------------------------------------------------------------------------
__global__ __launch_bounds__(256) void ln64(
    const float* __restrict__ x, const float* __restrict__ w,
    const float* __restrict__ b, _Float16* __restrict__ out, int M) {
  int lane = lane_id();
  int row = (int)((blockIdx.x * blockDim.x + threadIdx.x) >> 5);
  if (row >= M) return;
  const float* xr = x + (size_t)row * 64;
  float2 v = *(const float2*)(xr + lane * 2);
  float s = v.x + v.y;
#pragma unroll
  for (int off = 16; off; off >>= 1) s += __shfl_xor(s, off, 32);
  float mean = s * (1.f / 64.f);
  float dx = v.x - mean, dy = v.y - mean;
  float q = dx * dx + dy * dy;
#pragma unroll
  for (int off = 16; off; off >>= 1) q += __shfl_xor(q, off, 32);
  float rstd = rsqrtf(q * (1.f / 64.f) + 1e-5f);
  float2 wv = *(const float2*)(w + lane * 2);
  float2 bv = *(const float2*)(b + lane * 2);
  _Float16* orow = out + (size_t)row * 64 + lane * 2;
  orow[0] = (_Float16)(dx * rstd * wv.x + bv.x);
  orow[1] = (_Float16)(dy * rstd * wv.y + bv.y);
}

// ---------------------------------------------------------------------------
// Token GEMM: out[M,N] = A[M,K] @ W[N,K]^T + bias (+resid)
// Weights staged once per block into LDS; each wave runs MT 16-row tiles with
// the A fragments of tile mt+1 double-buffered ahead of tile mt's wmmas.
// If the whole B operand fits (NT*KC <= 8 fragments), hoist it to registers.
// ---------------------------------------------------------------------------
template <int NT, int KC, int MT, bool RESID, bool OUTF16>
__global__ __launch_bounds__(256) void gemm_tok(
    const _Float16* __restrict__ A, const _Float16* __restrict__ W,
    const float* __restrict__ bias, const float* __restrict__ resid,
    float* __restrict__ outf, _Float16* __restrict__ outh, int M) {
  constexpr int K = KC * 32, N = NT * 16;
  constexpr int WELEMS = N * K;                    // halves
  __shared__ _Float16 wlds[WELEMS];

  // ---- cooperative weight staging: global -> LDS, b128 chunks ----
  {
    constexpr int ITERS = WELEMS / (256 * 8);
#pragma unroll
    for (int i = 0; i < ITERS; ++i) {
      int idx = (i * 256 + (int)threadIdx.x) * 8;
      *(h8*)(wlds + idx) = *(const h8*)(W + idx);
    }
    __syncthreads();
  }

  int lane = lane_id();
  int wv = (int)(threadIdx.x >> 5);
  int nl = lane & 15;
  int mo = (lane & 16) ? 8 : 0;
  long tile0 = ((long)blockIdx.x * 8 + wv) * MT;

  constexpr bool HOIST = (NT * KC) <= 8;
  v16h bh[HOIST ? NT * KC : 1];
  if constexpr (HOIST) {
#pragma unroll
    for (int kc = 0; kc < KC; ++kc)
#pragma unroll
      for (int t = 0; t < NT; ++t)
        bh[kc * NT + t] = load_b_frag(wlds + (t * 16 + nl) * K + kc * 32);
  }

  // A-fragment double buffer across M tiles
  v16h aB[2][KC];
  {
    long m0 = tile0 * 16;
    if (m0 < M) {
      const _Float16* arow = A + (m0 + nl) * K;
#pragma unroll
      for (int kc = 0; kc < KC; ++kc) aB[0][kc] = load_a_frag(arow + kc * 32);
    }
  }

#pragma unroll
  for (int mt = 0; mt < MT; ++mt) {
    long m0 = (tile0 + mt) * 16;
    if (m0 >= M) continue;
    // prefetch next tile's A fragments into the other buffer
    if (mt + 1 < MT) {
      long m1 = (tile0 + mt + 1) * 16;
      if (m1 < M) {
        const _Float16* arow = A + (m1 + nl) * K;
#pragma unroll
        for (int kc = 0; kc < KC; ++kc)
          aB[(mt + 1) & 1][kc] = load_a_frag(arow + kc * 32);
      }
    }

    v8f acc[NT];
#pragma unroll
    for (int t = 0; t < NT; ++t) acc[t] = zero8();

#pragma unroll
    for (int kc = 0; kc < KC; ++kc) {
      v16h a = aB[mt & 1][kc];
      if constexpr (HOIST) {
#pragma unroll
        for (int t = 0; t < NT; ++t)
          acc[t] = wmma16(a, bh[kc * NT + t], acc[t]);
      } else {
#pragma unroll
        for (int tg = 0; tg < NT; tg += 4) {
          v16h bt[4];
#pragma unroll
          for (int u = 0; u < 4; ++u)
            bt[u] = load_b_frag(wlds + ((tg + u) * 16 + nl) * K + kc * 32);
#pragma unroll
          for (int u = 0; u < 4; ++u)
            acc[tg + u] = wmma16(a, bt[u], acc[tg + u]);
        }
      }
    }

#pragma unroll
    for (int t = 0; t < NT; ++t) {
      int n = t * 16 + nl;
      float bv = bias[n];
#pragma unroll
      for (int r = 0; r < 8; ++r) {
        long m = m0 + mo + r;
        float v = acc[t][r] + bv;
        if (RESID) v += resid[m * N + n];
        if (OUTF16) outh[m * N + n] = (_Float16)v;
        else        outf[m * N + n] = v;
      }
    }
  }
}

// ---------------------------------------------------------------------------
// Spatial-reduction conv as GEMM: M=2048 patches, K=4096 (reordered), N=64.
// 128 K-chunks, explicit 2-deep software pipeline: while wmmas for chunk k
// run, the fragments for chunk k+1 are already in flight.
// ---------------------------------------------------------------------------
__global__ __launch_bounds__(128) void sr_conv_gemm(
    const _Float16* __restrict__ xnh, const _Float16* __restrict__ srwT,
    const float* __restrict__ sr_b, float* __restrict__ xr_pre) {
  int lane = lane_id();
  int gw = (int)((blockIdx.x * blockDim.x + threadIdx.x) >> 5);
  int m0 = gw * 16;                    // patch tile base, M=2048
  int nl = lane & 15;
  int mrow = m0 + nl;
  int b = mrow >> 8, pi = mrow & 255, ph = pi >> 4, pw = pi & 15;
  const _Float16* rowpix =
      xnh + ((size_t)b * 16384 + (size_t)(ph * 8) * 128 + (size_t)(pw * 8)) * 64;

  v8f acc[4];
#pragma unroll
  for (int t = 0; t < 4; ++t) acc[t] = zero8();

  // chunk kc in [0,128): spatial tap s = kc>>1, half = kc&1
  auto load_chunk = [&](int kc, v16h& a, v16h* bt) {
    int s = kc >> 1, i = s >> 3, j = s & 7;
    a = load_a_frag(rowpix + (size_t)(i * 128 + j) * 64 + (kc & 1) * 32);
    int k0 = kc * 32;
#pragma unroll
    for (int t = 0; t < 4; ++t)
      bt[t] = load_b_frag(srwT + (size_t)(t * 16 + nl) * 4096 + k0);
  };

  v16h a0, a1, b0[4], b1[4];
  load_chunk(0, a0, b0);
  for (int kc = 0; kc < 128; kc += 2) {
    load_chunk(kc + 1, a1, b1);                 // in flight during wmmas below
#pragma unroll
    for (int t = 0; t < 4; ++t) acc[t] = wmma16(a0, b0[t], acc[t]);
    if (kc + 2 < 128) load_chunk(kc + 2, a0, b0);
#pragma unroll
    for (int t = 0; t < 4; ++t) acc[t] = wmma16(a1, b1[t], acc[t]);
  }

  int mo = (lane & 16) ? 8 : 0;
#pragma unroll
  for (int t = 0; t < 4; ++t) {
    int n = t * 16 + nl;
    float bv = sr_b[n];
#pragma unroll
    for (int r = 0; r < 8; ++r)
      xr_pre[(size_t)(m0 + mo + r) * 64 + n] = acc[t][r] + bv;
  }
}

// ---------------------------------------------------------------------------
// KV GEMM: M=2048, K=64, N=128. n<64 -> K matrix [B,256,64];
// n>=64 -> V stored transposed as vT [B,64,256] for the AV GEMM.
// ---------------------------------------------------------------------------
__global__ __launch_bounds__(256) void kv_gemm(
    const _Float16* __restrict__ xrh, const _Float16* __restrict__ kvw,
    const float* __restrict__ kv_b, _Float16* __restrict__ kbuf,
    _Float16* __restrict__ vT) {
  int lane = lane_id();
  int gw = (int)((blockIdx.x * blockDim.x + threadIdx.x) >> 5);
  int m0 = gw * 16;
  int nl = lane & 15;
  const _Float16* arow = xrh + (size_t)(m0 + nl) * 64;

  v8f acc[8];
#pragma unroll
  for (int t = 0; t < 8; ++t) acc[t] = zero8();
#pragma unroll
  for (int kc = 0; kc < 2; ++kc) {
    v16h a = load_a_frag(arow + kc * 32);
#pragma unroll
    for (int tg = 0; tg < 8; tg += 4) {
      v16h bt[4];
#pragma unroll
      for (int u = 0; u < 4; ++u)
        bt[u] = load_b_frag(kvw + (size_t)((tg + u) * 16 + nl) * 64 + kc * 32);
#pragma unroll
      for (int u = 0; u < 4; ++u)
        acc[tg + u] = wmma16(a, bt[u], acc[tg + u]);
    }
  }

  int mo = (lane & 16) ? 8 : 0;
#pragma unroll
  for (int t = 0; t < 8; ++t) {
    int n = t * 16 + nl;
    float bv = kv_b[n];
#pragma unroll
    for (int r = 0; r < 8; ++r) {
      int m = m0 + mo + r;
      float v = acc[t][r] + bv;
      int b = m >> 8, p = m & 255;
      if (n < 64) kbuf[(size_t)m * 64 + n] = (_Float16)v;
      else        vT[((size_t)b * 64 + (n - 64)) * 256 + p] = (_Float16)v;
    }
  }
}

// ---------------------------------------------------------------------------
// Attention: per wave 16 queries x all 256 keys. QK^T (WMMA) -> in-register
// softmax (butterfly shfl over the 16-lane half-group that owns each row)
// -> probs to LDS (reshape to A-fragments) -> AV (WMMA).
// ---------------------------------------------------------------------------
__global__ __launch_bounds__(128) void attn_kernel(
    const _Float16* __restrict__ qh, const _Float16* __restrict__ kbuf,
    const _Float16* __restrict__ vT, _Float16* __restrict__ obuf) {
  __shared__ _Float16 sprob[4][16 * 256];   // 32 KB, per-wave private slabs
  int lane = lane_id();
  int w = (int)(threadIdx.x >> 5);
  int qbase = blockIdx.x * 64 + w * 16;     // 16 query rows per wave
  int b = qbase >> 14;
  int nl = lane & 15;
  const _Float16* qrow = qh + (size_t)(qbase + nl) * 64;
  const _Float16* kb_ = kbuf + (size_t)b * 256 * 64;

  // ---- scores = q @ k^T, 16 x 256, B fragments double-buffered ----
  v8f sc[16];
#pragma unroll
  for (int t = 0; t < 16; ++t) sc[t] = zero8();
#pragma unroll
  for (int kc = 0; kc < 2; ++kc) {
    v16h a = load_a_frag(qrow + kc * 32);
    v16h c0[4], c1[4];
#pragma unroll
    for (int u = 0; u < 4; ++u)
      c0[u] = load_b_frag(kb_ + (size_t)(u * 16 + nl) * 64 + kc * 32);
#pragma unroll
    for (int tg = 0; tg < 16; tg += 8) {
#pragma unroll
      for (int u = 0; u < 4; ++u)
        c1[u] = load_b_frag(kb_ + (size_t)((tg + 4 + u) * 16 + nl) * 64 + kc * 32);
#pragma unroll
      for (int u = 0; u < 4; ++u) sc[tg + u] = wmma16(a, c0[u], sc[tg + u]);
      if (tg + 8 < 16) {
#pragma unroll
        for (int u = 0; u < 4; ++u)
          c0[u] = load_b_frag(kb_ + (size_t)((tg + 8 + u) * 16 + nl) * 64 + kc * 32);
      }
#pragma unroll
      for (int u = 0; u < 4; ++u) sc[tg + 4 + u] = wmma16(a, c1[u], sc[tg + 4 + u]);
    }
  }

  // ---- softmax over 256 keys, fully in registers ----
  const float scale = 0.125f;               // dh^-0.5, dh=64
  float mx[8], sm[8];
#pragma unroll
  for (int r = 0; r < 8; ++r) mx[r] = -1e30f;
#pragma unroll
  for (int t = 0; t < 16; ++t)
#pragma unroll
    for (int r = 0; r < 8; ++r) {
      float v = sc[t][r] * scale;
      sc[t][r] = v;
      mx[r] = fmaxf(mx[r], v);
    }
#pragma unroll
  for (int r = 0; r < 8; ++r) {
#pragma unroll
    for (int off = 1; off < 16; off <<= 1)
      mx[r] = fmaxf(mx[r], __shfl_xor(mx[r], off, 32));
    sm[r] = 0.f;
  }
#pragma unroll
  for (int t = 0; t < 16; ++t)
#pragma unroll
    for (int r = 0; r < 8; ++r) {
      float e = __expf(sc[t][r] - mx[r]);
      sc[t][r] = e;
      sm[r] += e;
    }
#pragma unroll
  for (int r = 0; r < 8; ++r) {
#pragma unroll
    for (int off = 1; off < 16; off <<= 1)
      sm[r] += __shfl_xor(sm[r], off, 32);
    sm[r] = 1.f / sm[r];
  }

  // ---- probs -> LDS (row-major 16x256 f16), same-wave in-order DS ----
  int mo = (lane & 16) ? 8 : 0;
  _Float16* sp = sprob[w];
#pragma unroll
  for (int t = 0; t < 16; ++t)
#pragma unroll
    for (int r = 0; r < 8; ++r)
      sp[(mo + r) * 256 + t * 16 + nl] = (_Float16)(sc[t][r] * sm[r]);

  // ---- o = probs @ v  (A from LDS, B from vT global, double-buffered) ----
  v8f o[4];
#pragma unroll
  for (int t = 0; t < 4; ++t) o[t] = zero8();
  const _Float16* vtb = vT + (size_t)b * 64 * 256;
  v16h a0, a1, d0[4], d1[4];
  a0 = load_a_frag(sp + (size_t)nl * 256);
#pragma unroll
  for (int t = 0; t < 4; ++t)
    d0[t] = load_b_frag(vtb + (size_t)(t * 16 + nl) * 256);
  for (int kc = 0; kc < 8; kc += 2) {
    a1 = load_a_frag(sp + (size_t)nl * 256 + (kc + 1) * 32);
#pragma unroll
    for (int t = 0; t < 4; ++t)
      d1[t] = load_b_frag(vtb + (size_t)(t * 16 + nl) * 256 + (kc + 1) * 32);
#pragma unroll
    for (int t = 0; t < 4; ++t) o[t] = wmma16(a0, d0[t], o[t]);
    if (kc + 2 < 8) {
      a0 = load_a_frag(sp + (size_t)nl * 256 + (kc + 2) * 32);
#pragma unroll
      for (int t = 0; t < 4; ++t)
        d0[t] = load_b_frag(vtb + (size_t)(t * 16 + nl) * 256 + (kc + 2) * 32);
    }
#pragma unroll
    for (int t = 0; t < 4; ++t) o[t] = wmma16(a1, d1[t], o[t]);
  }
#pragma unroll
  for (int t = 0; t < 4; ++t)
#pragma unroll
    for (int r = 0; r < 8; ++r)
      obuf[(size_t)(qbase + mo + r) * 64 + t * 16 + nl] = (_Float16)o[t][r];
}

// ---------------------------------------------------------------------------
// Depthwise 3x3 conv (channels innermost) + bias + exact GELU
// thread = one pixel x 8 channels
// ---------------------------------------------------------------------------
__global__ __launch_bounds__(256) void dwconv_gelu(
    const _Float16* __restrict__ h, const _Float16* __restrict__ dww,
    const float* __restrict__ dwb, _Float16* __restrict__ out) {
  int tid = (int)(blockIdx.x * blockDim.x + threadIdx.x);
  int cg = tid & 31;            // channel group (8 ch)
  int pix = tid >> 5;           // b*16384 + y*128 + x
  int x = pix & 127, y = (pix >> 7) & 127;
  int c0 = cg * 8;
  float acc[8];
#pragma unroll
  for (int c = 0; c < 8; ++c) acc[c] = dwb[c0 + c];
#pragma unroll
  for (int dy = -1; dy <= 1; ++dy) {
    int yy = y + dy;
    if (yy < 0 || yy > 127) continue;
#pragma unroll
    for (int dx = -1; dx <= 1; ++dx) {
      int xx = x + dx;
      if (xx < 0 || xx > 127) continue;
      h8 v = *(const h8*)(h + ((size_t)pix + dy * 128 + dx) * 256 + c0);
      int tap = (dy + 1) * 3 + (dx + 1);
#pragma unroll
      for (int c = 0; c < 8; ++c)
        acc[c] += (float)v[c] * (float)dww[(c0 + c) * 9 + tap];
    }
  }
  h8 o;
#pragma unroll
  for (int c = 0; c < 8; ++c) {
    float t = acc[c];
    o[c] = (_Float16)(0.5f * t * (1.f + erff(t * 0.70710678118f)));
  }
  *(h8*)(out + (size_t)pix * 256 + c0) = o;
}

// ---------------------------------------------------------------------------
// Host launcher
// ---------------------------------------------------------------------------
extern "C" void kernel_launch(void* const* d_in, const int* in_sizes, int n_in,
                              void* d_out, int out_size, void* d_ws, size_t ws_size,
                              hipStream_t stream) {
  (void)in_sizes; (void)n_in; (void)out_size; (void)ws_size;
  const float* x      = (const float*)d_in[0];
  const float* n1_w   = (const float*)d_in[3];
  const float* n1_b   = (const float*)d_in[4];
  const float* q_w    = (const float*)d_in[5];
  const float* q_b    = (const float*)d_in[6];
  const float* kv_w   = (const float*)d_in[7];
  const float* kv_b   = (const float*)d_in[8];
  const float* sr_w   = (const float*)d_in[9];
  const float* sr_b   = (const float*)d_in[10];
  const float* srn_w  = (const float*)d_in[11];
  const float* srn_b  = (const float*)d_in[12];
  const float* proj_w = (const float*)d_in[13];
  const float* proj_b = (const float*)d_in[14];
  const float* n2_w   = (const float*)d_in[15];
  const float* n2_b   = (const float*)d_in[16];
  const float* fc1_w  = (const float*)d_in[17];
  const float* fc1_b  = (const float*)d_in[18];
  const float* dw_w   = (const float*)d_in[19];
  const float* dw_b   = (const float*)d_in[20];
  const float* fc2_w  = (const float*)d_in[21];
  const float* fc2_b  = (const float*)d_in[22];
  float* out = (float*)d_out;

  const long M = 8L * 16384;   // 131072 tokens

  char* ws = (char*)d_ws;
  size_t off = 0;
  auto alloc = [&](size_t bytes) -> char* {
    off = (off + 255) & ~(size_t)255;
    char* p = ws + off;
    off += bytes;
    return p;
  };

  _Float16* xnh  = (_Float16*)alloc((size_t)M * 64 * 2);   // LN1 out (reused as obuf)
  _Float16* qh   = (_Float16*)alloc((size_t)M * 64 * 2);   // q (reused as xn2h)
  float*    x1   = (float*)   alloc((size_t)M * 64 * 4);   // residual stream
  _Float16* hbuf = (_Float16*)alloc((size_t)M * 256 * 2);  // fc1 out
  _Float16* h2   = (_Float16*)alloc((size_t)M * 256 * 2);  // dwconv+gelu out
  float*    xrp  = (float*)   alloc(2048 * 64 * 4);        // sr conv out (pre-LN)
  _Float16* xrh  = (_Float16*)alloc(2048 * 64 * 2);        // sr LN out
  _Float16* kbuf = (_Float16*)alloc(2048 * 64 * 2);        // K [B,256,64]
  _Float16* vT   = (_Float16*)alloc(2048 * 64 * 2);        // V^T [B,64,256]
  _Float16* qwh  = (_Float16*)alloc(4096 * 2);
  _Float16* kvwh = (_Float16*)alloc(8192 * 2);
  _Float16* pwh  = (_Float16*)alloc(4096 * 2);
  _Float16* f1wh = (_Float16*)alloc(16384 * 2);
  _Float16* f2wh = (_Float16*)alloc(16384 * 2);
  _Float16* srwh = (_Float16*)alloc(262144 * 2);
  _Float16* dwwh = (_Float16*)alloc(2304 * 2);
  _Float16* obuf = xnh;    // xnh dead after q-gemm + sr-conv
  _Float16* xn2h = qh;     // qh dead after attention

  // --- weights to f16 (sr_w also K-reordered) ---
  cvt_f16<<<16, 256, 0, stream>>>(q_w,   qwh,  4096);
  cvt_f16<<<32, 256, 0, stream>>>(kv_w,  kvwh, 8192);
  cvt_f16<<<16, 256, 0, stream>>>(proj_w, pwh, 4096);
  cvt_f16<<<64, 256, 0, stream>>>(fc1_w, f1wh, 16384);
  cvt_f16<<<64, 256, 0, stream>>>(fc2_w, f2wh, 16384);
  cvt_f16<<<9,  256, 0, stream>>>(dw_w,  dwwh, 2304);
  srw_reorder<<<1024, 256, 0, stream>>>(sr_w, srwh);

  // Each gemm_tok block: 8 waves x MT=4 tiles x 16 rows = 512 rows.
  const int GB = (int)(M / 512);   // 256 blocks

  // --- attention branch ---
  ln64<<<16384, 256, 0, stream>>>(x, n1_w, n1_b, xnh, (int)M);
  gemm_tok<4, 2, 4, false, true><<<GB, 256, 0, stream>>>(
      xnh, qwh, q_b, nullptr, nullptr, qh, (int)M);
  sr_conv_gemm<<<32, 128, 0, stream>>>(xnh, srwh, sr_b, xrp);
  ln64<<<256, 256, 0, stream>>>(xrp, srn_w, srn_b, xrh, 2048);
  kv_gemm<<<16, 256, 0, stream>>>(xrh, kvwh, kv_b, kbuf, vT);
  attn_kernel<<<2048, 128, 0, stream>>>(qh, kbuf, vT, obuf);
  gemm_tok<4, 2, 4, true, false><<<GB, 256, 0, stream>>>(
      obuf, pwh, proj_b, x, x1, nullptr, (int)M);

  // --- Mix-FFN branch ---
  ln64<<<16384, 256, 0, stream>>>(x1, n2_w, n2_b, xn2h, (int)M);
  gemm_tok<16, 2, 4, false, true><<<GB, 256, 0, stream>>>(
      xn2h, f1wh, fc1_b, nullptr, nullptr, hbuf, (int)M);
  dwconv_gelu<<<16384, 256, 0, stream>>>(hbuf, dwwh, dw_b, h2);
  gemm_tok<4, 8, 4, true, false><<<GB, 256, 0, stream>>>(
      h2, f2wh, fc2_b, x1, out, nullptr, (int)M);
}
